// NonLocalTarget_51977694216737
// MI455X (gfx1250) — compile-verified, matrix operand order checked by reading, not run on hardware
//
#include <hip/hip_runtime.h>

#define Tt   4
#define Bb   8
#define Cc   256
#define CI   128
#define HWs  1024
#define Nn   4096
#define KEEP 410
#define INVK (1.0f / 410.0f)

typedef __attribute__((ext_vector_type(16))) _Float16 v16h;
typedef __attribute__((ext_vector_type(8)))  _Float16 v8h;
typedef __attribute__((ext_vector_type(8)))  float    v8f;

#define WMMA_F16(a, b, c) \
  __builtin_amdgcn_wmma_f32_16x16x32_f16(false, (a), false, (b), (short)0, (c), false, false)

// ---------------- workspace layout (bytes) ----------------
static constexpr size_t SZ_W    = 32768;                    // 128*256 (=256*128) halves
static constexpr size_t SZ_BNC  = (size_t)Bb * Nn * CI;     // 4,194,304 halves
static constexpr size_t OFF_WG  = 0;
static constexpr size_t OFF_WTH = OFF_WG  + SZ_W * 2;
static constexpr size_t OFF_WPH = OFF_WTH + SZ_W * 2;
static constexpr size_t OFF_WWW = OFF_WPH + SZ_W * 2;
static constexpr size_t OFF_TH  = OFF_WWW + SZ_W * 2;       // theta_s  f16 [b][n][o]
static constexpr size_t OFF_PH  = OFF_TH  + SZ_BNC * 2;     // phi_t^T  f16 [b][n][o]
static constexpr size_t OFF_GTT = OFF_PH  + SZ_BNC * 2;     // g_t^T    f16 [b][o][n]
static constexpr size_t OFF_YH  = OFF_GTT + SZ_BNC * 2;     // y        f16 [b][n][o]
static constexpr size_t OFF_WY  = OFF_YH  + SZ_BNC * 2;     // Wy       f32 [b][n][c]
static constexpr size_t OFF_ST  = OFF_WY  + (size_t)Bb * Nn * Cc * 4;
static constexpr size_t WS_NEED = OFF_ST + 1024 * 4;

// d_out regions (floats): z | phi_t | g_t
static constexpr size_t OUT_PHI = 8388608;
static constexpr size_t OUT_G   = 12582912;

// ---------------- fragment helpers ----------------
// A / B^T fragment load from row-major [row][k] storage (stride in halves,
// multiple of 8). ISA 16-bit A layout: lane l<16 holds M=l, K={k0..k0+7,
// k0+16..k0+23}; lane l>=16 holds M=l-16, K shifted by +8.
__device__ __forceinline__ v16h load_frag(const _Float16* base, int row, int stride, int k0) {
  int l  = (int)(threadIdx.x & 31u);
  int m  = l & 15;
  int kk = k0 + ((l >> 4) << 3);
  const _Float16* p = base + (size_t)(row + m) * (size_t)stride + kk;
  v8h lo = *(const v8h*)(p);
  v8h hi = *(const v8h*)(p + 16);
  v16h r;
#pragma unroll
  for (int i = 0; i < 8; ++i) { r[i] = lo[i]; r[i + 8] = hi[i]; }
  return r;
}

__device__ __forceinline__ unsigned short okey(unsigned short u) {
  return (unsigned short)(u ^ ((u & 0x8000u) ? 0xFFFFu : 0x8000u));
}

__device__ __forceinline__ _Float16 maskscale(_Float16 h, unsigned short thr, float sc) {
  unsigned short u = __builtin_bit_cast(unsigned short, h);
  if (okey(u) < thr) return (_Float16)0.0f;
  return (_Float16)((float)h * sc);
}

// ---------------- kernels ----------------
__global__ __launch_bounds__(256) void k_init(
    const float* gw, const float* thw, const float* phw, const float* Ww,
    _Float16* hgw, _Float16* hthw, _Float16* hphw, _Float16* hWw, float* stats) {
  int i = blockIdx.x * 256 + threadIdx.x;
  if (i < (int)SZ_W) {
    hgw[i]  = (_Float16)gw[i];
    hthw[i] = (_Float16)thw[i];
    hphw[i] = (_Float16)phw[i];
    hWw[i]  = (_Float16)Ww[i];
  }
  if (i < 1024) stats[i] = 0.0f;
}

// Projections: for proj p (0:g on tf, 1:theta on sf, 2:phi on tf):
// out[n][o] = sum_c src[b][c][n] * W[o][c] + b[o].  M=16 n-rows per block,
// 8 waves cover o=0..127, K=256.
__global__ __launch_bounds__(256) void k_proj(
    const float* targets, const float* test,
    const _Float16* hgw, const _Float16* hthw, const _Float16* hphw,
    const float* gb, const float* thb, const float* phb,
    float* out, _Float16* ws_th, _Float16* ws_ph, _Float16* ws_gtT) {
  __shared__ _Float16 ldsA[16 * 256];   // A tile [m][c], 8 KB
  int tid = threadIdx.x, wv = tid >> 5;
  int n0 = blockIdx.x * 16, b = blockIdx.y, p = blockIdx.z;
  const float* src = (p == 1) ? test : targets;
  int t = n0 >> 10, hwb = n0 & 1023;
  size_t base = (((size_t)t * Bb + b) * Cc) * HWs + hwb;
  for (int idx = tid; idx < 16 * 256; idx += 256) {
    int k = idx >> 4, m = idx & 15;
    ldsA[m * 256 + k] = (_Float16)src[base + (size_t)k * HWs + m];
  }
  __syncthreads();
  const _Float16* wsel = (p == 0) ? hgw : (p == 1) ? hthw : hphw;
  const float*    bsel = (p == 0) ? gb  : (p == 1) ? thb  : phb;
  int o0 = wv * 16;
  v8f acc = {};
#pragma unroll
  for (int kc = 0; kc < 8; ++kc) {
    v16h a   = load_frag(ldsA, 0, 256, kc * 32);
    v16h bfr = load_frag(wsel, o0, 256, kc * 32);
    acc = WMMA_F16(a, bfr, acc);
  }
  int l = tid & 31, j = l & 15, mh = (l >> 4) * 8;
  float bias = bsel[o0 + j];
  if (p == 0) {
    float* outG = out + OUT_G;
#pragma unroll
    for (int r = 0; r < 8; ++r) {
      int n = n0 + mh + r, o = o0 + j;
      float v = acc[r] + bias;
      outG[((size_t)b * Nn + n) * CI + o]  = v;               // g_t (b,Nt,Ci) f32 output
      ws_gtT[((size_t)b * CI + o) * Nn + n] = (_Float16)v;    // g_t^T f16 (B operand)
    }
  } else if (p == 1) {
#pragma unroll
    for (int r = 0; r < 8; ++r) {
      int n = n0 + mh + r, o = o0 + j;
      ws_th[((size_t)b * Nn + n) * CI + o] = (_Float16)(acc[r] + bias);
    }
  } else {
    float* outP = out + OUT_PHI;
#pragma unroll
    for (int r = 0; r < 8; ++r) {
      int n = n0 + mh + r, o = o0 + j;
      float v = acc[r] + bias;
      outP[((size_t)b * CI + o) * Nn + n]  = v;               // phi_t (b,Ci,Nt) f32 output
      ws_ph[((size_t)b * Nn + n) * CI + o] = (_Float16)v;     // phi_t^T f16 (B operand)
    }
  }
}

// Fused: f = theta_s x phi_t (16 x 4096 row-block kept in LDS, f16),
// exact per-row top-k threshold via 2-pass radix select, then
// y = (mask(f)/keep) x g_t via WMMA.  4 waves / 128 threads.
__global__ __launch_bounds__(128) void k_attn(
    const _Float16* ws_th, const _Float16* ws_ph, const _Float16* ws_gtT,
    _Float16* ws_yh) {
  __shared__ _Float16 fbuf[16 * Nn];        // 128 KB (CDNA5 320KB LDS/WGP)
  __shared__ unsigned int hist[16 * 256];   // 16 KB
  __shared__ int s_hi[16], s_rem[16];
  __shared__ unsigned short s_thr[16];
  int tid = threadIdx.x, wv = tid >> 5;
  int s0 = blockIdx.x * 16, b = blockIdx.y;
  const _Float16* thA = ws_th  + (size_t)b * Nn * CI;
  const _Float16* phB = ws_ph  + (size_t)b * Nn * CI;
  const _Float16* gB  = ws_gtT + (size_t)b * CI * Nn;
  int l = tid & 31, j = l & 15, mh = (l >> 4) * 8;

  // ---- phase 1: f row-block, K=128 ----
  v16h aTh[4];
#pragma unroll
  for (int kc = 0; kc < 4; ++kc) aTh[kc] = load_frag(thA, s0, CI, kc * 32);
  for (int i = wv; i < Nn / 16; i += 4) {
    int n0 = i * 16;
    v8f acc = {};
#pragma unroll
    for (int kc = 0; kc < 4; ++kc) {
      v16h bfr = load_frag(phB, n0, CI, kc * 32);
      acc = WMMA_F16(aTh[kc], bfr, acc);
    }
#pragma unroll
    for (int r = 0; r < 8; ++r) fbuf[(mh + r) * Nn + n0 + j] = (_Float16)acc[r];
  }
  __syncthreads();

  // ---- phase 2: exact 410th-largest (over f16 keys) per row ----
  int row = tid & 15, sub = tid >> 4;            // 8 threads per row
  const unsigned short* fb = (const unsigned short*)fbuf;
  for (int pass = 0; pass < 2; ++pass) {
    for (int i = tid; i < 16 * 256; i += 128) hist[i] = 0u;
    __syncthreads();
    int nb = sub * (Nn / 8), ne = nb + Nn / 8;
    if (pass == 0) {
      for (int n = nb; n < ne; ++n) {
        unsigned short k = okey(fb[row * Nn + n]);
        atomicAdd(&hist[row * 256 + (k >> 8)], 1u);
      }
    } else {
      int hi = s_hi[row];
      for (int n = nb; n < ne; ++n) {
        unsigned short k = okey(fb[row * Nn + n]);
        if ((int)(k >> 8) == hi) atomicAdd(&hist[row * 256 + (k & 0xFF)], 1u);
      }
    }
    __syncthreads();
    if (sub == 0) {
      const unsigned int* h = &hist[row * 256];
      int need = (pass == 0) ? KEEP : s_rem[row];
      int d = 255;
      for (;;) { int c = (int)h[d]; if (need - c <= 0 || d == 0) break; need -= c; --d; }
      if (pass == 0) { s_hi[row] = d; s_rem[row] = need; }
      else           s_thr[row] = (unsigned short)((s_hi[row] << 8) | d);
    }
    __syncthreads();
  }

  // ---- phase 3: y = mask(f)*invK x g_t, wave -> 2 o-tiles, K=4096 ----
  unsigned short thr = s_thr[l & 15];
  int oT0 = (2 * wv) * 16, oT1 = (2 * wv + 1) * 16;
  v8f acc0 = {}, acc1 = {};
  for (int nc = 0; nc < Nn / 32; ++nc) {
    int n0 = nc * 32;
    int kk = (l >> 4) << 3;
    const _Float16* p = fbuf + (l & 15) * Nn + n0 + kk;
    v16h a;
#pragma unroll
    for (int i2 = 0; i2 < 8; ++i2) {
      a[i2]     = maskscale(p[i2],      thr, INVK);
      a[i2 + 8] = maskscale(p[i2 + 16], thr, INVK);
    }
    v16h b0 = load_frag(gB, oT0, Nn, n0);
    v16h b1 = load_frag(gB, oT1, Nn, n0);
    acc0 = WMMA_F16(a, b0, acc0);
    acc1 = WMMA_F16(a, b1, acc1);
  }
#pragma unroll
  for (int r = 0; r < 8; ++r) {
    size_t sidx = (size_t)b * Nn + s0 + mh + r;
    ws_yh[sidx * CI + oT0 + j] = (_Float16)acc0[r];
    ws_yh[sidx * CI + oT1 + j] = (_Float16)acc1[r];
  }
}

// Wy[b][n][o] = sum_i Ww[o][i]*y[b][n][i] + Wb[o]; accumulate BN sum/sumsq.
__global__ __launch_bounds__(128) void k_W(
    const _Float16* ws_yh, const _Float16* hWw, const float* Wb,
    float* ws_Wy, float* stats) {
  int tid = threadIdx.x, wv = tid >> 5;
  int n0 = blockIdx.x * 16, b = blockIdx.y;
  const _Float16* A = ws_yh + (size_t)b * Nn * CI;
  v16h af[4];
#pragma unroll
  for (int kc = 0; kc < 4; ++kc) af[kc] = load_frag(A, n0, CI, kc * 32);
  int l = tid & 31, j = l & 15, mh = (l >> 4) * 8;
  for (int ot = 0; ot < 4; ++ot) {
    int o0 = (wv * 4 + ot) * 16;
    v8f acc = {};
#pragma unroll
    for (int kc = 0; kc < 4; ++kc) {
      v16h bfr = load_frag(hWw, o0, CI, kc * 32);
      acc = WMMA_F16(af[kc], bfr, acc);
    }
    float bias = Wb[o0 + j];
    float s1 = 0.f, s2 = 0.f;
#pragma unroll
    for (int r = 0; r < 8; ++r) {
      float v = acc[r] + bias;
      ws_Wy[((size_t)b * Nn + n0 + mh + r) * Cc + o0 + j] = v;
      s1 += v; s2 += v * v;
    }
    atomicAdd(&stats[o0 + j], s1);
    atomicAdd(&stats[256 + o0 + j], s2);
  }
}

__global__ void k_bn(const float* stats, const float* gamma, const float* beta, float* sc) {
  int c = threadIdx.x;
  if (c < 256) {
    float inv  = 1.0f / (float)((size_t)Bb * Nn);
    float mean = stats[c] * inv;
    float var  = stats[256 + c] * inv - mean * mean;
    float rs   = rsqrtf(var + 1e-5f);
    float scale = rs * gamma[c];
    sc[c]       = scale;
    sc[256 + c] = beta[c] - mean * scale;
  }
}

// z[t,b,c,h,w] = BN(Wy) + test_feat ; output flat (t*b, c, h, w) == input flat order.
__global__ __launch_bounds__(256) void k_out(
    const float* ws_Wy, const float* sc, const float* test, float* out) {
  size_t idx = (size_t)blockIdx.x * 256 + threadIdx.x;   // < 8,388,608
  int hw = (int)(idx & 1023);
  int c  = (int)((idx >> 10) & 255);
  int tb = (int)(idx >> 18);
  int t = tb >> 3, b = tb & 7;
  size_t wyi = ((size_t)b * Nn + (size_t)t * HWs + hw) * Cc + c;
  out[idx] = ws_Wy[wyi] * sc[c] + sc[256 + c] + test[idx];
}

// ---------------- launch ----------------
extern "C" void kernel_launch(void* const* d_in, const int* in_sizes, int n_in,
                              void* d_out, int out_size, void* d_ws, size_t ws_size,
                              hipStream_t stream) {
  (void)in_sizes; (void)n_in; (void)out_size;
  if (ws_size < WS_NEED) return;
  const float* targets = (const float*)d_in[0];
  const float* test    = (const float*)d_in[1];
  const float* gw      = (const float*)d_in[2];
  const float* gb      = (const float*)d_in[3];
  const float* thw     = (const float*)d_in[4];
  const float* thb     = (const float*)d_in[5];
  const float* phw     = (const float*)d_in[6];
  const float* phb     = (const float*)d_in[7];
  const float* Ww      = (const float*)d_in[8];
  const float* Wb      = (const float*)d_in[9];
  const float* gamma   = (const float*)d_in[10];
  const float* beta    = (const float*)d_in[11];

  char* ws = (char*)d_ws;
  _Float16* hgw   = (_Float16*)(ws + OFF_WG);
  _Float16* hthw  = (_Float16*)(ws + OFF_WTH);
  _Float16* hphw  = (_Float16*)(ws + OFF_WPH);
  _Float16* hWw   = (_Float16*)(ws + OFF_WWW);
  _Float16* wsTh  = (_Float16*)(ws + OFF_TH);
  _Float16* wsPh  = (_Float16*)(ws + OFF_PH);
  _Float16* wsGtT = (_Float16*)(ws + OFF_GTT);
  _Float16* wsYh  = (_Float16*)(ws + OFF_YH);
  float*    wsWy  = (float*)(ws + OFF_WY);
  float*    stats = (float*)(ws + OFF_ST);
  float*    out   = (float*)d_out;

  k_init<<<128, 256, 0, stream>>>(gw, thw, phw, Ww, hgw, hthw, hphw, hWw, stats);
  k_proj<<<dim3(Nn / 16, Bb, 3), 256, 0, stream>>>(targets, test, hgw, hthw, hphw,
                                                   gb, thb, phb, out, wsTh, wsPh, wsGtT);
  k_attn<<<dim3(Nn / 16, Bb), 128, 0, stream>>>(wsTh, wsPh, wsGtT, wsYh);
  k_W<<<dim3(Nn / 16, Bb), 128, 0, stream>>>(wsYh, hWw, Wb, wsWy, stats);
  k_bn<<<1, 256, 0, stream>>>(stats, gamma, beta, stats + 512);
  k_out<<<32768, 256, 0, stream>>>(wsWy, stats + 512, test, out);
}
